// DepthModule_4982162063835
// MI455X (gfx1250) — matrix operand order repaired and checked by reading, not compiled, not run on hardware
//
#include <hip/hip_runtime.h>
#include <hip/hip_bf16.h>

typedef _Float16 h16;
typedef __attribute__((ext_vector_type(16))) _Float16 v16h;
typedef __attribute__((ext_vector_type(8)))  _Float16 v8h;
typedef __attribute__((ext_vector_type(8)))  float    v8f;
typedef __attribute__((ext_vector_type(4)))  float    v4f;

// ---- CDNA5 WMMA 16x16x32 f16 lane->element maps (CDNA5 ISA 7.12.2) ----
// A (16x32, MxK): lane row m = lane&15; half h -> K = (lane>>4)*16 + h  (contiguous order chosen by this code, not the ISA table order)
// B (32x16, KxN): column n = lane&15; K = (lane>>4)*16 + h  (contiguous order chosen by this code, not the ISA table order)
// C/D (16x16 f32, v8f): N = lane&15; M = r + 8*(lane>>4)  (8 consecutive M rows per lane)
__device__ __forceinline__ int a_kidx(int lane, int h) {
    return h + 16 * (lane >> 4);
}

// =====================================================================
// Zero-fill (for padded activation buffers). n8 = #elements / 8.
// =====================================================================
__global__ void zero_h16(h16* __restrict__ p, long n8) {
    long i = (long)blockIdx.x * blockDim.x + threadIdx.x;
    if (i < n8) {
        v8h z = {};
        *(v8h*)(p + i * 8) = z;
    }
}

// =====================================================================
// Weight packing into A-fragment lane layout.
// pack2d: w [32][32][3][3] f32 -> packed [kidx9][tile t][lane][16] f16  (9216)
// =====================================================================
__global__ void pack_w2d(const float* __restrict__ w, h16* __restrict__ out) {
    int i = blockIdx.x * blockDim.x + threadIdx.x;
    if (i >= 9216) return;
    const int h    = i & 15;
    const int L    = (i >> 4) & 31;
    const int t    = (i >> 9) & 1;
    const int kidx = i >> 10;                 // 0..8
    const int o = t * 16 + (L & 15);
    const int c = a_kidx(L, h);
    out[i] = (h16)w[(o * 32 + c) * 9 + kidx];
}

// pack3d: w [32][64][3][3][3] f32 -> packed [kidx27][chunk][tile][lane][16] f16 (55296)
__global__ void pack_w3d(const float* __restrict__ w, h16* __restrict__ out) {
    int i = blockIdx.x * blockDim.x + threadIdx.x;
    if (i >= 55296) return;
    const int h     = i & 15;
    const int L     = (i >> 4) & 31;
    const int t     = (i >> 9) & 1;
    const int chunk = (i >> 10) & 1;
    const int kidx  = i >> 11;                // 0..26
    const int o = t * 16 + (L & 15);
    const int c = chunk * 32 + a_kidx(L, h);
    out[i] = (h16)w[(o * 64 + c) * 27 + kidx];
}

// =====================================================================
// conv1: 3->32, 3x3, stride 2, SAME (pad_lo=0), + normalize + ReLU
// img [5][3][480][640] f32 -> out padded NHWC [5][241][321][32] f16 (interior)
// =====================================================================
__global__ void conv1_kernel(const float* __restrict__ img, const float* __restrict__ w1,
                             const float* __restrict__ b1, h16* __restrict__ out) {
    __shared__ float sw[32 * 3 * 9];
    __shared__ float sb[32];
    for (int i = threadIdx.x; i < 864; i += blockDim.x) sw[i] = w1[i];
    if (threadIdx.x < 32) sb[threadIdx.x] = b1[threadIdx.x];
    __syncthreads();

    int idx = blockIdx.x * blockDim.x + threadIdx.x;   // 5*240*320 = 384000
    if (idx >= 384000) return;
    const int ox = idx % 320;
    int r        = idx / 320;
    const int oy = r % 240;
    const int f  = r / 240;

    float s[32];
#pragma unroll
    for (int o = 0; o < 32; ++o) s[o] = sb[o];

#pragma unroll
    for (int ky = 0; ky < 3; ++ky) {
        const int iy = 2 * oy + ky;
        if (iy >= 480) continue;
#pragma unroll
        for (int kx = 0; kx < 3; ++kx) {
            const int ix = 2 * ox + kx;
            if (ix >= 640) continue;
#pragma unroll
            for (int c = 0; c < 3; ++c) {
                float v = img[(((size_t)f * 3 + c) * 480 + iy) * 640 + ix];
                v = 2.0f * (v * (1.0f / 255.0f)) - 1.0f;
#pragma unroll
                for (int o = 0; o < 32; ++o) s[o] += v * sw[(o * 3 + c) * 9 + ky * 3 + kx];
            }
        }
    }
    h16* op = out + (((size_t)f * 241 + oy) * 321 + ox) * 32;
#pragma unroll
    for (int q = 0; q < 4; ++q) {
        v8h p;
#pragma unroll
        for (int j = 0; j < 8; ++j) {
            float v = s[q * 8 + j];
            p[j] = (h16)(v > 0.0f ? v : 0.0f);
        }
        *(v8h*)(op + q * 8) = p;
    }
}

// =====================================================================
// conv 3x3 stride 2, SAME (pad_lo=0), 32->32, + ReLU, via WMMA.
// in padded NHWC [F][HinP][WinP][32] f16 (zero halo at high edges),
// out padded NHWC [F][HoutP][WoutP][*] (interior [Hout][Wout]).
// One wave: 16 output pixels x 32 output channels; 18 WMMA, zero branches.
// =====================================================================
template <typename OutT>
__global__ void conv2d_s2_wmma(const h16* __restrict__ in, const h16* __restrict__ wp,
                               const float* __restrict__ bias, OutT* __restrict__ out,
                               int F, int HinP, int WinP, int Hout, int Wout,
                               int HoutP, int WoutP) {
    __shared__ __align__(32) h16 sw[9216];       // packed A fragments, 18 KB
    for (int i = threadIdx.x; i < 9216; i += blockDim.x) sw[i] = wp[i];
    __syncthreads();

    const int lane   = threadIdx.x & 31;
    const int wave   = threadIdx.x >> 5;
    const int tilesX = Wout >> 4;
    const int tile   = blockIdx.x * (blockDim.x >> 5) + wave;

    const int tx = tile % tilesX;
    int rem      = tile / tilesX;
    const int oy = rem % Hout;
    const int f  = rem / Hout;
    const int x0 = tx << 4;

    const int n = lane & 15;
    const int g = lane >> 4;

    v8f acc0, acc1;
#pragma unroll
    for (int r = 0; r < 8; ++r) {
        acc0[r] = bias[r + 8 * g];
        acc1[r] = bias[16 + r + 8 * g];
    }

    const int ix0 = 2 * (x0 + n);
#pragma unroll
    for (int ky = 0; ky < 3; ++ky) {
        const int iy = 2 * oy + ky;              // <= HinP-1 (zero halo)
        const h16* rowp = in + (((size_t)f * HinP + iy) * WinP) * 32 + (g << 4);
#pragma unroll
        for (int kx = 0; kx < 3; ++kx) {
            const int kidx = ky * 3 + kx;
            const v16h b  = *(const v16h*)(rowp + (size_t)(ix0 + kx) * 32);
            const v16h a0 = *(const v16h*)(&sw[((kidx * 2 + 0) * 32 + lane) * 16]);
            const v16h a1 = *(const v16h*)(&sw[((kidx * 2 + 1) * 32 + lane) * 16]);
            acc0 = __builtin_amdgcn_wmma_f32_16x16x32_f16(false, a0, false, b, (short)0, acc0, false, false);
            acc1 = __builtin_amdgcn_wmma_f32_16x16x32_f16(false, a1, false, b, (short)0, acc1, false, false);
        }
    }

    OutT* op = out + ((((size_t)f * HoutP + oy) * WoutP + x0 + n) << 5);
    if constexpr (sizeof(OutT) == 2) {
        v8h p0, p1;
#pragma unroll
        for (int r = 0; r < 8; ++r) {
            p0[r] = (h16)(acc0[r] > 0.0f ? acc0[r] : 0.0f);
            p1[r] = (h16)(acc1[r] > 0.0f ? acc1[r] : 0.0f);
        }
        *(v8h*)((h16*)op + 8 * g)      = p0;
        *(v8h*)((h16*)op + 16 + 8 * g) = p1;
    } else {
        v4f q;
#pragma unroll
        for (int r = 0; r < 4; ++r) q[r] = acc0[r] > 0.0f ? acc0[r] : 0.0f;
        *(v4f*)((float*)op + 8 * g) = q;
#pragma unroll
        for (int r = 0; r < 4; ++r) q[r] = acc0[r + 4] > 0.0f ? acc0[r + 4] : 0.0f;
        *(v4f*)((float*)op + 8 * g + 4) = q;
#pragma unroll
        for (int r = 0; r < 4; ++r) q[r] = acc1[r] > 0.0f ? acc1[r] : 0.0f;
        *(v4f*)((float*)op + 16 + 8 * g) = q;
#pragma unroll
        for (int r = 0; r < 4; ++r) q[r] = acc1[r + 4] > 0.0f ? acc1[r + 4] : 0.0f;
        *(v4f*)((float*)op + 16 + 8 * g + 4) = q;
    }
}

// =====================================================================
// Plane-sweep cost volume.
// fmap3 NHWC [5][60][80][32] f32 -> vol padded NDHWC [34][62][82][64] f16
// (logical voxel (d,y,x) at physical (d+1, y+1, x+1); halo pre-zeroed)
// =====================================================================
__global__ void build_volume(const float* __restrict__ poses, const float* __restrict__ intr,
                             const float* __restrict__ fmap3, h16* __restrict__ vol) {
    int idx = blockIdx.x * blockDim.x + threadIdx.x;
    if (idx >= 32 * 60 * 80) return;
    const int x = idx % 80;
    const int y = (idx / 80) % 60;
    const int d = idx / 4800;

    const float fx = intr[0] * 0.25f, fy = intr[1] * 0.25f;
    const float cx = intr[2] * 0.25f, cy = intr[3] * 0.25f;
    const float depth = 0.25f + (float)d * (7.75f / 31.0f);

    const float px = depth * ((float)x - cx) / fx;
    const float py = depth * ((float)y - cy) / fy;
    const float pz = depth;

    float R0[3][3], t0v[3];
#pragma unroll
    for (int r = 0; r < 3; ++r) {
#pragma unroll
        for (int c = 0; c < 3; ++c) R0[r][c] = poses[r * 4 + c];
        t0v[r] = poses[r * 4 + 3];
    }

    float acc[32];
#pragma unroll
    for (int c = 0; c < 32; ++c) acc[c] = 0.0f;

    for (int f = 0; f < 5; ++f) {
        const float* Pf = poses + f * 16;
        float RG[3][3], tG[3];
#pragma unroll
        for (int r = 0; r < 3; ++r)
#pragma unroll
            for (int c = 0; c < 3; ++c)
                RG[r][c] = Pf[r * 4 + 0] * R0[c][0] + Pf[r * 4 + 1] * R0[c][1] + Pf[r * 4 + 2] * R0[c][2];
#pragma unroll
        for (int r = 0; r < 3; ++r)
            tG[r] = Pf[r * 4 + 3] - (RG[r][0] * t0v[0] + RG[r][1] * t0v[1] + RG[r][2] * t0v[2]);

        const float X0 = RG[0][0] * px + RG[0][1] * py + RG[0][2] * pz + tG[0];
        const float X1 = RG[1][0] * px + RG[1][1] * py + RG[1][2] * pz + tG[1];
        const float X2 = RG[2][0] * px + RG[2][1] * py + RG[2][2] * pz + tG[2];

        const float u = (fx * X0 + cx * X2) / (X2 + 1e-8f);
        const float v = (fy * X1 + cy * X2) / (X2 + 1e-8f);

        const float xf = floorf(u), yf = floorf(v);
        const float wx = u - xf, wy = v - yf;
        const int x0i = (int)xf, y0i = (int)yf;
        const bool valid = (u >= 0.0f) && (u <= 79.0f) && (v >= 0.0f) && (v <= 59.0f);
        if (!valid) continue;
        const int xa = min(max(x0i, 0), 79), xb = min(max(x0i + 1, 0), 79);
        const int ya = min(max(y0i, 0), 59), yb = min(max(y0i + 1, 0), 59);
        const float w00 = (1.0f - wx) * (1.0f - wy), w10 = wx * (1.0f - wy);
        const float w01 = (1.0f - wx) * wy,          w11 = wx * wy;

        const float* fm  = fmap3 + (size_t)f * 60 * 80 * 32;
        const float* p00 = fm + (ya * 80 + xa) * 32;
        const float* p10 = fm + (ya * 80 + xb) * 32;
        const float* p01 = fm + (yb * 80 + xa) * 32;
        const float* p11 = fm + (yb * 80 + xb) * 32;
#pragma unroll
        for (int c = 0; c < 32; ++c)
            acc[c] += p00[c] * w00 + p10[c] * w10 + p01[c] * w01 + p11[c] * w11;
    }

    h16* vp = vol + ((((size_t)(d + 1) * 62 + (y + 1)) * 82 + (x + 1)) << 6);
    const float* f0 = fmap3 + (size_t)(y * 80 + x) * 32;
#pragma unroll
    for (int q = 0; q < 4; ++q) {
        v8h pr, pa;
#pragma unroll
        for (int j = 0; j < 8; ++j) {
            pr[j] = (h16)f0[q * 8 + j];
            pa[j] = (h16)(acc[q * 8 + j] * 0.2f);
        }
        *(v8h*)(vp + q * 8)      = pr;
        *(v8h*)(vp + 32 + q * 8) = pa;
    }
}

// =====================================================================
// conv3d: 64->32, 3x3x3, SAME, + ReLU, via WMMA.  Zero branches (halo).
// vol padded NDHWC [34][62][82][64] f16, out NDHWC [32][60][80][32] f16
// One wave: 16 voxels x 32 output channels; 108 WMMA per tile.
// =====================================================================
__global__ void conv3d_wmma(const h16* __restrict__ vol, const h16* __restrict__ wp,
                            const float* __restrict__ bias, h16* __restrict__ out) {
    __shared__ __align__(32) h16 sw[18432];      // one kd slice of packed fragments, 36 KB

    const int lane = threadIdx.x & 31;
    const int wave = threadIdx.x >> 5;
    const int tile = blockIdx.x * (blockDim.x >> 5) + wave;   // 32*60*5 tiles

    const int tx = tile % 5;
    int rem      = tile / 5;
    const int y  = rem % 60;
    const int d  = rem / 60;
    const int x0 = tx << 4;

    const int n = lane & 15;
    const int g = lane >> 4;

    v8f acc0, acc1;
#pragma unroll
    for (int r = 0; r < 8; ++r) {
        acc0[r] = bias[r + 8 * g];
        acc1[r] = bias[16 + r + 8 * g];
    }

    for (int kd = 0; kd < 3; ++kd) {
        __syncthreads();
        for (int i = threadIdx.x; i < 18432; i += blockDim.x) sw[i] = wp[kd * 18432 + i];
        __syncthreads();
        if (kd < 2) __builtin_prefetch(wp + (kd + 1) * 18432, 0, 1);   // global_prefetch_b8

        const int pd = d + kd;                    // physical depth (halo +1)
#pragma unroll
        for (int ky = 0; ky < 3; ++ky) {
            const int py = y + ky;                // physical row
            const h16* rowp = vol + ((((size_t)pd * 62 + py) * 82 + x0 + n) << 6) + (g << 4);
#pragma unroll
            for (int kx = 0; kx < 3; ++kx) {
                const int kidx9 = ky * 3 + kx;
#pragma unroll
                for (int ch = 0; ch < 2; ++ch) {
                    const v16h b  = *(const v16h*)(rowp + ((size_t)kx << 6) + ch * 32);
                    const v16h a0 = *(const v16h*)(&sw[(((kidx9 * 2 + ch) * 2 + 0) << 9) + lane * 16]);
                    const v16h a1 = *(const v16h*)(&sw[(((kidx9 * 2 + ch) * 2 + 1) << 9) + lane * 16]);
                    acc0 = __builtin_amdgcn_wmma_f32_16x16x32_f16(false, a0, false, b, (short)0, acc0, false, false);
                    acc1 = __builtin_amdgcn_wmma_f32_16x16x32_f16(false, a1, false, b, (short)0, acc1, false, false);
                }
            }
        }
    }

    h16* op = out + (((size_t)(d * 60 + y) * 80 + x0 + n) << 5);
    v8h p0, p1;
#pragma unroll
    for (int r = 0; r < 8; ++r) {
        p0[r] = (h16)(acc0[r] > 0.0f ? acc0[r] : 0.0f);
        p1[r] = (h16)(acc1[r] > 0.0f ? acc1[r] : 0.0f);
    }
    *(v8h*)(op + 8 * g)      = p0;
    *(v8h*)(op + 16 + 8 * g) = p1;
}

// =====================================================================
// Final: conv3d 32->1 (3x3x3, SAME) + softmax over depth + expectation.
// h3 NDHWC [32][60][80][32] f16 -> out [60][80] f32
// =====================================================================
__global__ void depth_regress(const h16* __restrict__ h3, const float* __restrict__ wd2,
                              const float* __restrict__ bd2, float* __restrict__ out) {
    __shared__ float swt[27 * 32];               // transposed: [kidx][c]
    for (int i = threadIdx.x; i < 864; i += blockDim.x) swt[(i % 27) * 32 + i / 27] = wd2[i];
    __syncthreads();

    int idx = blockIdx.x * blockDim.x + threadIdx.x;
    if (idx >= 4800) return;
    const int x = idx % 80;
    const int y = idx / 80;

    const float bb = bd2[0];
    float logit[32];
    for (int d = 0; d < 32; ++d) {
        float s = bb;
#pragma unroll
        for (int kd = 0; kd < 3; ++kd) {
            const int id = d + kd - 1;
            if (id < 0 || id >= 32) continue;
#pragma unroll
            for (int ky = 0; ky < 3; ++ky) {
                const int iy = y + ky - 1;
                if (iy < 0 || iy >= 60) continue;
#pragma unroll
                for (int kx = 0; kx < 3; ++kx) {
                    const int ix = x + kx - 1;
                    if (ix < 0 || ix >= 80) continue;
                    const h16*   hp = h3 + (((size_t)(id * 60 + iy) * 80 + ix) << 5);
                    const float* wq = &swt[(kd * 9 + ky * 3 + kx) * 32];
#pragma unroll
                    for (int q = 0; q < 4; ++q) {
                        const v8h hv = *(const v8h*)(hp + q * 8);
#pragma unroll
                        for (int j = 0; j < 8; ++j) s += (float)hv[j] * wq[q * 8 + j];
                    }
                }
            }
        }
        logit[d] = s;
    }

    float mx = logit[0];
#pragma unroll
    for (int d = 1; d < 32; ++d) mx = fmaxf(mx, logit[d]);
    float se = 0.0f, ev = 0.0f;
#pragma unroll
    for (int d = 0; d < 32; ++d) {
        const float e = expf(logit[d] - mx);
        se += e;
        ev += e * (0.25f + (float)d * (7.75f / 31.0f));
    }
    out[idx] = ev / se;
}

// =====================================================================
extern "C" void kernel_launch(void* const* d_in, const int* in_sizes, int n_in,
                              void* d_out, int out_size, void* d_ws, size_t ws_size,
                              hipStream_t stream) {
    const float* poses  = (const float*)d_in[0];
    const float* images = (const float*)d_in[1];
    const float* intr   = (const float*)d_in[2];
    const float* w1     = (const float*)d_in[3];
    const float* b1     = (const float*)d_in[4];
    const float* w2     = (const float*)d_in[5];
    const float* b2     = (const float*)d_in[6];
    const float* w3     = (const float*)d_in[7];
    const float* b3     = (const float*)d_in[8];
    const float* wd1    = (const float*)d_in[9];
    const float* bd1    = (const float*)d_in[10];
    const float* wd2    = (const float*)d_in[11];
    const float* bd2    = (const float*)d_in[12];
    float* out = (float*)d_out;

    char* base = (char*)d_ws;
    size_t off = 0;
    auto alloc = [&](size_t bytes) -> void* {
        void* p = base + off;
        off = (off + bytes + 255) & ~(size_t)255;
        return p;
    };
    const size_t n_f1  = (size_t)5 * 241 * 321 * 32;   // 12,377,760
    const size_t n_f2  = (size_t)5 * 121 * 161 * 32;   //  3,116,960
    const size_t n_vol = (size_t)34 * 62 * 82 * 64;    // 11,062,784
    h16*   fmap1 = (h16*)  alloc(n_f1 * 2);            // padded NHWC
    h16*   fmap2 = (h16*)  alloc(n_f2 * 2);            // padded NHWC
    float* fmap3 = (float*)alloc((size_t)5 * 60 * 80 * 32 * 4);   // NHWC f32
    h16*   vol   = (h16*)  alloc(n_vol * 2);           // padded NDHWC
    h16*   h3    = (h16*)  alloc((size_t)32 * 60 * 80 * 32 * 2);  // NDHWC
    h16*   w2p   = (h16*)  alloc((size_t)9216 * 2);
    h16*   w3p   = (h16*)  alloc((size_t)9216 * 2);
    h16*   wd1p  = (h16*)  alloc((size_t)55296 * 2);

    // zero padded buffers (halo must read as 0)
    zero_h16<<<(int)((n_f1 / 8 + 255) / 256), 256, 0, stream>>>(fmap1, (long)(n_f1 / 8));
    zero_h16<<<(int)((n_f2 / 8 + 255) / 256), 256, 0, stream>>>(fmap2, (long)(n_f2 / 8));
    zero_h16<<<(int)((n_vol / 8 + 255) / 256), 256, 0, stream>>>(vol, (long)(n_vol / 8));

    pack_w2d<<<36, 256, 0, stream>>>(w2, w2p);
    pack_w2d<<<36, 256, 0, stream>>>(w3, w3p);
    pack_w3d<<<216, 256, 0, stream>>>(wd1, wd1p);

    // conv1: one thread per output pixel (all 32 channels): 5*240*320
    conv1_kernel<<<1500, 256, 0, stream>>>(images, w1, b1, fmap1);

    // conv2: 5*120*(160/16)=6000 wave-tiles, 4 waves/block
    conv2d_s2_wmma<h16><<<1500, 128, 0, stream>>>(fmap1, w2p, b2, fmap2,
                                                  5, 241, 321, 120, 160, 121, 161);

    // conv3: 5*60*(80/16)=1500 wave-tiles (output fmap3 unpadded)
    conv2d_s2_wmma<float><<<375, 128, 0, stream>>>(fmap2, w3p, b3, fmap3,
                                                   5, 121, 161, 60, 80, 60, 80);

    // cost volume: 32*60*80 threads
    build_volume<<<600, 256, 0, stream>>>(poses, intr, fmap3, vol);

    // conv3d 64->32: 32*60*(80/16)=9600 wave-tiles
    conv3d_wmma<<<2400, 128, 0, stream>>>(vol, wd1p, bd1, h3);

    // logits + softmax + expectation: 4800 threads
    depth_regress<<<(4800 + 127) / 128, 128, 0, stream>>>(h3, wd2, bd2, out);
}
